// MultiRelationalSelfAttention_69363721831025
// MI455X (gfx1250) — matrix operand order repaired and checked by reading, not compiled, not run on hardware
//
#include <hip/hip_runtime.h>
#include <hip/hip_bf16.h>

// CDNA5 / gfx1250 fused multi-relational attention.
// d_out = [O (B,H,N,D) | attn_weight (B,H,N,N)], both f32.

typedef __attribute__((ext_vector_type(16))) __bf16 v16bf;
typedef __attribute__((ext_vector_type(8)))  float  v8f;
typedef __attribute__((ext_vector_type(4)))  float  v4f;

constexpr int Bb = 4, Hh = 16, Nn = 1024, Dd = 64;
constexpr int ROWS   = 16;   // query rows per workgroup
constexpr int NWAVES = 4;    // waves per workgroup (wave32)
constexpr float INV_SCALE = 0.125f;  // 1/sqrt(64)

__global__ __launch_bounds__(NWAVES * 32)
void mrsa_fused_kernel(const float* __restrict__ Q, const float* __restrict__ K,
                       const float* __restrict__ V, const float* __restrict__ A,
                       const float* __restrict__ wE,
                       float* __restrict__ outO, float* __restrict__ outP)
{
    __shared__ float s_row[ROWS * Nn];  // 64 KB score/prob tile (of 320 KB/WGP)

    const int lane = threadIdx.x & 31;
    const int wave = threadIdx.x >> 5;
    const int row_base = blockIdx.x * ROWS;
    const int h = blockIdx.y;
    const int b = blockIdx.z;

    const size_t bh  = (size_t)b * Hh + h;
    const float* Qbh = Q + bh * Nn * Dd;
    const float* Kbh = K + bh * Nn * Dd;
    const float* Vbh = V + bh * Nn * Dd;

    // per-edge-type head weights (T=4)
    const float w0 = wE[0 * Hh + h], w1 = wE[1 * Hh + h];
    const float w2 = wE[2 * Hh + h], w3 = wE[3 * Hh + h];

    const int lmod   = lane & 15;
    const int hi     = (lane >> 4) & 1;  // lane half
    const int khalfA = hi * 8;           // A-fragment K base / 2 (contiguous k = 16*hi + i)
    const int koffB  = hi * 16;          // B-fragment K base (contiguous k = 16*hi + i, same as A)

    // ---- Q tile as two bf16 A-fragments (D = 64 = 2 x K32) ----
    v16bf aQ0, aQ1;
    {
        const float* qrow = Qbh + (size_t)(row_base + lmod) * Dd;
        #pragma unroll
        for (int i = 0; i < 16; ++i) {
            const int f = khalfA * 2 + i;
            aQ0[i] = (__bf16)qrow[f];
            aQ1[i] = (__bf16)qrow[32 + f];
        }
    }

    const size_t tstr  = (size_t)Bb * Nn * Nn;           // A stride between edge types
    const float* Arow0 = A + (size_t)b * Nn * Nn;        // A[0][b]

    // ---- pass 1: biased scores, waves stripe the 64 column tiles ----
    for (int jt = wave; jt < Nn / 16; jt += NWAVES) {
        const int col_base = jt * 16;

        if (jt + NWAVES < Nn / 16)  // hint next K tile -> global_prefetch_b8
            __builtin_prefetch(Kbh + (size_t)(col_base + NWAVES * 16 + lmod) * Dd, 0, 3);

        v16bf bK0, bK1;
        const float* krow = Kbh + (size_t)(col_base + lmod) * Dd;
        #pragma unroll
        for (int i = 0; i < 16; ++i) {
            bK0[i] = (__bf16)krow[koffB + i];
            bK1[i] = (__bf16)krow[32 + koffB + i];
        }

        v8f acc = {};
        acc = __builtin_amdgcn_wmma_f32_16x16x32_bf16(false, aQ0, false, bK0,
                                                      (short)0, acc, false, false);
        acc = __builtin_amdgcn_wmma_f32_16x16x32_bf16(false, aQ1, false, bK1,
                                                      (short)0, acc, false, false);

        const int col_g = col_base + lmod;
        #pragma unroll
        for (int r = 0; r < 8; ++r) {
            const int row_l = r + hi * 8;                 // C layout: VGPR r -> row r / r+8
            const int row_g = row_base + row_l;
            const float* ap = Arow0 + (size_t)row_g * Nn + col_g;
            const float bias = w0 * ap[0] + w1 * ap[tstr]
                             + w2 * ap[2 * tstr] + w3 * ap[3 * tstr];
            s_row[row_l * Nn + col_g] = acc[r] * (1.0f + bias) * INV_SCALE;
        }
    }

    __syncthreads();

    // ---- pass 2: row softmax + NT-stream attn_weight (268 MB write) ----
    // float4 lanes: 1024 floats/row = 256 v4f = 8 iterations of 32 lanes.
    #pragma unroll
    for (int rr = 0; rr < ROWS / NWAVES; ++rr) {
        const int row_l = wave * (ROWS / NWAVES) + rr;
        v4f* srow4 = (v4f*)(s_row + row_l * Nn);

        float mx = -3.4e38f;
        #pragma unroll 2
        for (int j = lane; j < Nn / 4; j += 32) {
            const v4f v = srow4[j];
            mx = fmaxf(mx, fmaxf(fmaxf(v.x, v.y), fmaxf(v.z, v.w)));
        }
        #pragma unroll
        for (int off = 16; off > 0; off >>= 1) mx = fmaxf(mx, __shfl_xor(mx, off, 32));

        float sum = 0.0f;
        #pragma unroll 2
        for (int j = lane; j < Nn / 4; j += 32) {
            v4f v = srow4[j];
            v.x = __expf(v.x - mx); v.y = __expf(v.y - mx);
            v.z = __expf(v.z - mx); v.w = __expf(v.w - mx);
            srow4[j] = v;
            sum += (v.x + v.y) + (v.z + v.w);
        }
        #pragma unroll
        for (int off = 16; off > 0; off >>= 1) sum += __shfl_xor(sum, off, 32);

        const float inv = 1.0f / sum;
        v4f* prow4 = (v4f*)(outP + (bh * Nn + (size_t)(row_base + row_l)) * Nn);
        #pragma unroll 2
        for (int j = lane; j < Nn / 4; j += 32) {
            v4f v = srow4[j];
            v.x *= inv; v.y *= inv; v.z *= inv; v.w *= inv;
            srow4[j] = v;                            // normalized probs stay for PV
            __builtin_nontemporal_store(v, prow4 + j);  // NT: don't evict A from L2
        }
    }

    __syncthreads();

    // ---- pass 3: O = P @ V, each wave owns a 16-wide slab of D ----
    const int dcol = wave * 16 + lmod;
    v8f o = {};
    for (int kt = 0; kt < Nn; kt += 32) {
        v16bf aP, bV;
        const float* srowA = s_row + lmod * Nn + kt;
        #pragma unroll
        for (int i = 0; i < 16; ++i) {
            const int f = khalfA * 2 + i;
            aP[i] = (__bf16)srowA[f];                              // ds_load gathers
            bV[i] = (__bf16)Vbh[(size_t)(kt + koffB + i) * Dd + dcol];
        }
        o = __builtin_amdgcn_wmma_f32_16x16x32_bf16(false, aP, false, bV,
                                                    (short)0, o, false, false);
    }

    float* orow = outO + (bh * Nn + (size_t)row_base) * Dd;
    #pragma unroll
    for (int r = 0; r < 8; ++r) {
        const int row_l = r + hi * 8;
        __builtin_nontemporal_store(o[r], orow + (size_t)row_l * Dd + dcol);
    }
}

extern "C" void kernel_launch(void* const* d_in, const int* in_sizes, int n_in,
                              void* d_out, int out_size, void* d_ws, size_t ws_size,
                              hipStream_t stream) {
    (void)in_sizes; (void)n_in; (void)out_size; (void)d_ws; (void)ws_size;
    const float* Q  = (const float*)d_in[0];
    const float* K  = (const float*)d_in[1];
    const float* V  = (const float*)d_in[2];
    const float* A  = (const float*)d_in[3];
    const float* wE = (const float*)d_in[4];

    float* outO = (float*)d_out;
    float* outP = outO + (size_t)Bb * Hh * Nn * Dd;  // attn_weight follows O

    dim3 grid(Nn / ROWS, Hh, Bb);   // (64, 16, 4) workgroups
    mrsa_fused_kernel<<<grid, NWAVES * 32, 0, stream>>>(Q, K, V, A, wE, outO, outP);
}
